// NodeModel_88923002897017
// MI455X (gfx1250) — matrix operand order repaired and checked by reading, not compiled, not run on hardware
//
#include <hip/hip_runtime.h>

// bf16 WMMA vector types (gfx1250, wave32)
typedef __attribute__((ext_vector_type(16))) __bf16 v16bf;
typedef __attribute__((ext_vector_type(8)))  __bf16 v8bf;
typedef __attribute__((ext_vector_type(8)))  float  v8f;
typedef __attribute__((ext_vector_type(4)))  float  v4f;   // native vec for NT loads

typedef union { v16bf v; v8bf h[2]; } v16bf_u;

#define TM 32          // rows (edges or nodes) per block tile
#define GN_EPS 1e-5f

// ---- WMMA fragment loaders -------------------------------------------------
// A: 16x32 bf16 tile from row-major LDS buffer (stride lda elements).
// Layout: lane m=lane&15, half=lane>>4; elems j=0..15 -> K = k0+half*16+j
// (matches B fragment). Two contiguous 16B chunks.
__device__ inline v16bf load_a_frag(const __bf16* base, int m0, int k0,
                                    int lane, int lda) {
    int m  = m0 + (lane & 15);
    int hh = lane >> 4;
    const __bf16* p = base + m * lda + k0 + hh * 16;
    v16bf_u u;
    u.h[0] = *(const v8bf*)p;
    u.h[1] = *(const v8bf*)(p + 8);
    return u.v;
}

// B: 32x16 bf16 tile; weights stored pre-transposed Wt[n][k] (K=256 stride) so
// lane n=lane&15, half=lane>>4 reads 16 contiguous K values = one 32B load.
__device__ inline v16bf load_b_frag(const __bf16* __restrict__ Wt, int n0,
                                    int k0, int lane) {
    int n  = n0 + (lane & 15);
    int hh = lane >> 4;
    return *(const v16bf*)(Wt + (size_t)n * 256 + k0 + hh * 16);
}

// C/D: f32 16x16; lane n=lane&15, elem r -> M = m0 + 8*(lane>>4) + r.
__device__ inline void store_tile_bias(float* H, const v8f& acc,
                                       const float* __restrict__ bias,
                                       int m0, int n0, int lane, int ldc) {
    int n  = n0 + (lane & 15);
    int hh = lane >> 4;
    float bv = bias[n];
#pragma unroll
    for (int r = 0; r < 8; ++r) H[(m0 + hh * 8 + r) * ldc + n] = acc[r] + bv;
}

// GEMM helper: one wave computes a 16 x (NT*16) strip of D = A[32x256] @ W.
// A fragments are loaded once per K-step and reused for all NT column tiles.
// mt = wave>>2 selects the M tile; nt = (wave&3) + j*4 selects column tiles.
template <int NT>
__device__ inline void gemm_strip(const __bf16* Abuf,
                                  const __bf16* __restrict__ Wt,
                                  float* Hbuf, const float* __restrict__ bias,
                                  int wave, int lane, int ldc) {
    const int m0 = (wave >> 2) * 16;
    const int ntb = wave & 3;
    v8f acc[NT];
#pragma unroll
    for (int j = 0; j < NT; ++j) acc[j] = (v8f){};
#pragma unroll
    for (int k0 = 0; k0 < 256; k0 += 32) {
        v16bf a = load_a_frag(Abuf, m0, k0, lane, 256);
#pragma unroll
        for (int j = 0; j < NT; ++j) {
            int n0 = (ntb + j * 4) * 16;
            v16bf b = load_b_frag(Wt, n0, k0, lane);
            acc[j] = __builtin_amdgcn_wmma_f32_16x16x32_bf16(
                false, a, false, b, (short)0, acc[j], false, false);
        }
    }
#pragma unroll
    for (int j = 0; j < NT; ++j)
        store_tile_bias(Hbuf, acc[j], bias, m0, (ntb + j * 4) * 16, lane, ldc);
}

// ---- helper kernels --------------------------------------------------------
__global__ void zero_f32_kernel(float* p, size_t n) {
    size_t i = (size_t)blockIdx.x * blockDim.x + threadIdx.x;
    size_t st = (size_t)gridDim.x * blockDim.x;
    for (; i < n; i += st) p[i] = 0.0f;
}

// dst[n*K + k] = (bf16) src[k*Nc + n]   (K rows, Nc cols in src)
__global__ void transpose_to_bf16_kernel(const float* __restrict__ src,
                                         __bf16* __restrict__ dst,
                                         int K, int Nc) {
    int i = blockIdx.x * blockDim.x + threadIdx.x;
    if (i < K * Nc) {
        int k = i / Nc, n = i - k * Nc;
        dst[(size_t)n * K + k] = (__bf16)src[i];
    }
}

// ---- fused edge kernel -----------------------------------------------------
// per tile of 32 edges: gather+concat -> GEMM1(256x256) -> GN+ReLU ->
// GEMM2(256x128) -> GN+ReLU -> atomic scatter-add (sums, cnt)
__global__ __launch_bounds__(256) void edge_kernel(
    const float* __restrict__ x, const int* __restrict__ ei,
    const float* __restrict__ ea,
    const __bf16* __restrict__ W1a_t, const float* __restrict__ b1a,
    const float* __restrict__ g1aw, const float* __restrict__ g1ab,
    const __bf16* __restrict__ W1b_t, const float* __restrict__ b1b,
    const float* __restrict__ g1bw, const float* __restrict__ g1bb,
    float* __restrict__ sums, float* __restrict__ cnt, long long E) {
    __shared__ __bf16 Abuf[TM * 256];   // 16 KB bf16 activations
    __shared__ float  Hbuf[TM * 256];   // 32 KB f32 GEMM results
    __shared__ int    rowIdx[TM];
    __shared__ int    colIdx[TM];

    const int t = threadIdx.x;
    const int lane = t & 31;
    const int wave = t >> 5;
    const long long e0 = (long long)blockIdx.x * TM;

    if (t < TM) {
        long long e = e0 + t;
        int r = 0, c = 0;
        if (e < E) { r = ei[e]; c = ei[E + e]; }
        rowIdx[t] = r;
        colIdx[t] = c;
    }
    __syncthreads();

    // stage concat([x[row], edge_attr[e]]) as bf16 into Abuf.
    // thread covers 32 contiguous channels of one row -> 8 vec4 loads.
    // edge_attr is a one-shot 410MB stream: non-temporal keeps L2 for
    // x / weights / scatter accumulators.
    {
        int r  = t >> 3;
        int c0 = (t & 7) * 32;
        long long e = e0 + r;
        bool ok = e < E;
        const v4f* src4 = (c0 < 128)
            ? (const v4f*)(x + (size_t)rowIdx[r] * 128 + c0)
            : (const v4f*)(ea + (size_t)e * 128 + (c0 - 128));
        bool nt = (c0 >= 128);
#pragma unroll
        for (int q = 0; q < 8; ++q) {
            v4f v = (v4f){0.f, 0.f, 0.f, 0.f};
            if (ok) v = nt ? __builtin_nontemporal_load(src4 + q) : src4[q];
            __bf16* dst = Abuf + r * 256 + c0 + q * 4;
            dst[0] = (__bf16)v.x; dst[1] = (__bf16)v.y;
            dst[2] = (__bf16)v.z; dst[3] = (__bf16)v.w;
        }
    }
    __syncthreads();

    // GEMM1: [32x256] @ W1a[256x256]; each wave: 1 M-tile x 4 N-tiles
    gemm_strip<4>(Abuf, W1a_t, Hbuf, b1a, wave, lane, 256);
    __syncthreads();

    // GroupNorm1 (C=256, 8 groups of 32) + ReLU -> Abuf (bf16)
    {
        int r = t >> 3, g = t & 7;
        const float* hp = Hbuf + r * 256 + g * 32;
        float s = 0.0f, s2 = 0.0f;
#pragma unroll
        for (int i = 0; i < 32; ++i) { float v = hp[i]; s += v; s2 += v * v; }
        float mu = s * (1.0f / 32.0f);
        float var = s2 * (1.0f / 32.0f) - mu * mu;
        float rs = __frsqrt_rn(var + GN_EPS);
#pragma unroll
        for (int i = 0; i < 32; ++i) {
            int c = g * 32 + i;
            float v = (hp[i] - mu) * rs * g1aw[c] + g1ab[c];
            Abuf[r * 256 + c] = (__bf16)fmaxf(v, 0.0f);
        }
    }
    __syncthreads();

    // GEMM2: [32x256] @ W1b[256x128]; each wave: 1 M-tile x 2 N-tiles
    gemm_strip<2>(Abuf, W1b_t, Hbuf, b1b, wave, lane, 128);
    __syncthreads();

    // GroupNorm2 (C=128, 8 groups of 16) + ReLU + scatter atomics
    {
        int r = t >> 3, g = t & 7;
        long long e = e0 + r;
        const float* hp = Hbuf + r * 128 + g * 16;
        float s = 0.0f, s2 = 0.0f;
#pragma unroll
        for (int i = 0; i < 16; ++i) { float v = hp[i]; s += v; s2 += v * v; }
        float mu = s * (1.0f / 16.0f);
        float var = s2 * (1.0f / 16.0f) - mu * mu;
        float rs = __frsqrt_rn(var + GN_EPS);
        if (e < E) {
            float* dst = sums + (size_t)colIdx[r] * 128 + g * 16;
#pragma unroll
            for (int i = 0; i < 16; ++i) {
                int c = g * 16 + i;
                float v = (hp[i] - mu) * rs * g1bw[c] + g1bb[c];
                atomicAdd(dst + i, fmaxf(v, 0.0f));
            }
        }
    }
    if (t < TM) {
        long long e = e0 + t;
        if (e < E) atomicAdd(cnt + colIdx[t], 1.0f);
    }
}

// ---- fused node kernel -----------------------------------------------------
__global__ __launch_bounds__(256) void node_kernel(
    const float* __restrict__ x, const float* __restrict__ sums,
    const float* __restrict__ cnt,
    const __bf16* __restrict__ W2a_t, const float* __restrict__ b2a,
    const float* __restrict__ g2aw, const float* __restrict__ g2ab,
    const __bf16* __restrict__ W2b_t, const float* __restrict__ b2b,
    float* __restrict__ out, int N) {
    __shared__ __bf16 Abuf[TM * 256];
    __shared__ float  Hbuf[TM * 256];

    const int t = threadIdx.x;
    const int lane = t & 31;
    const int wave = t >> 5;
    const int nd0 = blockIdx.x * TM;

    // stage concat([x[node], sums[node]/max(cnt,1)]) as bf16
    {
        int r  = t >> 3;
        int c0 = (t & 7) * 32;
        int node = nd0 + r;
        bool ok = node < N;
        float inv = ok ? (1.0f / fmaxf(cnt[node], 1.0f)) : 0.0f;
        bool isagg = (c0 >= 128);
        const v4f* src4 = isagg
            ? (const v4f*)(sums + (size_t)node * 128 + (c0 - 128))
            : (const v4f*)(x + (size_t)node * 128 + c0);
        float scale = isagg ? inv : 1.0f;
#pragma unroll
        for (int q = 0; q < 8; ++q) {
            v4f v = (v4f){0.f, 0.f, 0.f, 0.f};
            if (ok) v = src4[q];
            __bf16* dst = Abuf + r * 256 + c0 + q * 4;
            dst[0] = (__bf16)(v.x * scale); dst[1] = (__bf16)(v.y * scale);
            dst[2] = (__bf16)(v.z * scale); dst[3] = (__bf16)(v.w * scale);
        }
    }
    __syncthreads();

    // GEMM: [32x256] @ W2a[256x256]
    gemm_strip<4>(Abuf, W2a_t, Hbuf, b2a, wave, lane, 256);
    __syncthreads();

    // GroupNorm (C=256) + ReLU -> Abuf
    {
        int r = t >> 3, g = t & 7;
        const float* hp = Hbuf + r * 256 + g * 32;
        float s = 0.0f, s2 = 0.0f;
#pragma unroll
        for (int i = 0; i < 32; ++i) { float v = hp[i]; s += v; s2 += v * v; }
        float mu = s * (1.0f / 32.0f);
        float var = s2 * (1.0f / 32.0f) - mu * mu;
        float rs = __frsqrt_rn(var + GN_EPS);
#pragma unroll
        for (int i = 0; i < 32; ++i) {
            int c = g * 32 + i;
            float v = (hp[i] - mu) * rs * g2aw[c] + g2ab[c];
            Abuf[r * 256 + c] = (__bf16)fmaxf(v, 0.0f);
        }
    }
    __syncthreads();

    // GEMM: [32x256] @ W2b[256x128] + bias -> d_out (no GN/ReLU)
    {
        const int m0 = (wave >> 2) * 16;
        const int ntb = wave & 3;
        v8f acc[2];
        acc[0] = (v8f){}; acc[1] = (v8f){};
#pragma unroll
        for (int k0 = 0; k0 < 256; k0 += 32) {
            v16bf a = load_a_frag(Abuf, m0, k0, lane, 256);
#pragma unroll
            for (int j = 0; j < 2; ++j) {
                int n0 = (ntb + j * 4) * 16;
                v16bf b = load_b_frag(W2b_t, n0, k0, lane);
                acc[j] = __builtin_amdgcn_wmma_f32_16x16x32_bf16(
                    false, a, false, b, (short)0, acc[j], false, false);
            }
        }
        int hh = lane >> 4;
#pragma unroll
        for (int j = 0; j < 2; ++j) {
            int n = (ntb + j * 4) * 16 + (lane & 15);
            float bv = b2b[n];
#pragma unroll
            for (int r = 0; r < 8; ++r) {
                int node = nd0 + m0 + hh * 8 + r;
                if (node < N) out[(size_t)node * 128 + n] = acc[j][r] + bv;
            }
        }
    }
}

// ---- launch ---------------------------------------------------------------
extern "C" void kernel_launch(void* const* d_in, const int* in_sizes, int n_in,
                              void* d_out, int out_size, void* d_ws,
                              size_t ws_size, hipStream_t stream) {
    const float* x    = (const float*)d_in[0];
    const int*   ei   = (const int*)d_in[1];
    const float* ea   = (const float*)d_in[2];
    const float* W1a  = (const float*)d_in[3];
    const float* b1a  = (const float*)d_in[4];
    const float* g1aw = (const float*)d_in[5];
    const float* g1ab = (const float*)d_in[6];
    const float* W1b  = (const float*)d_in[7];
    const float* b1b  = (const float*)d_in[8];
    const float* g1bw = (const float*)d_in[9];
    const float* g1bb = (const float*)d_in[10];
    const float* W2a  = (const float*)d_in[11];
    const float* b2a  = (const float*)d_in[12];
    const float* g2aw = (const float*)d_in[13];
    const float* g2ab = (const float*)d_in[14];
    const float* W2b  = (const float*)d_in[15];
    const float* b2b  = (const float*)d_in[16];

    const int       N = in_sizes[0] / 128;
    const long long E = (long long)(in_sizes[2] / 128);

    // workspace layout: sums[N*128] f32 | cnt[N] f32 | bf16 transposed weights
    float*  sums  = (float*)d_ws;
    float*  cnt   = sums + (size_t)N * 128;
    __bf16* W1a_t = (__bf16*)(cnt + N);
    __bf16* W1b_t = W1a_t + 256 * 256;
    __bf16* W2a_t = W1b_t + 128 * 256;
    __bf16* W2b_t = W2a_t + 256 * 256;

    zero_f32_kernel<<<2048, 256, 0, stream>>>(sums, (size_t)N * 128 + N);
    transpose_to_bf16_kernel<<<(256 * 256 + 255) / 256, 256, 0, stream>>>(
        W1a, W1a_t, 256, 256);
    transpose_to_bf16_kernel<<<(256 * 128 + 255) / 256, 256, 0, stream>>>(
        W1b, W1b_t, 256, 128);
    transpose_to_bf16_kernel<<<(256 * 256 + 255) / 256, 256, 0, stream>>>(
        W2a, W2a_t, 256, 256);
    transpose_to_bf16_kernel<<<(256 * 128 + 255) / 256, 256, 0, stream>>>(
        W2b, W2b_t, 256, 128);

    const int nEB = (int)((E + TM - 1) / TM);
    edge_kernel<<<nEB, 256, 0, stream>>>(x, ei, ea, W1a_t, b1a, g1aw, g1ab,
                                         W1b_t, b1b, g1bw, g1bb, sums, cnt, E);

    const int nNB = (N + TM - 1) / TM;
    node_kernel<<<nNB, 256, 0, stream>>>(x, sums, cnt, W2a_t, b2a, g2aw, g2ab,
                                         W2b_t, b2b, (float*)d_out, N);
}
